// MultiHeadSelfAttention_67130338836805
// MI455X (gfx1250) — matrix operand order repaired and checked by reading, not compiled, not run on hardware
//
#include <hip/hip_runtime.h>

#define H 1024
#define NH 16
#define HD 64
#define SQ 2048
#define BATCH 2
#define M_TOT (BATCH * SQ) /* 4096 */

typedef __attribute__((ext_vector_type(16))) __bf16 v16bf;
typedef __attribute__((ext_vector_type(8))) float v8f;

union FragBF {
    v16bf bf;
    unsigned int w[8];
};

__device__ __forceinline__ unsigned short f2bf(float f) {
    unsigned int u = __builtin_bit_cast(unsigned int, f);
    unsigned int r = 0x7FFFu + ((u >> 16) & 1u);
    return (unsigned short)((u + r) >> 16);
}

__device__ __forceinline__ unsigned int pack2bf(float lo, float hi) {
    return (unsigned int)f2bf(lo) | ((unsigned int)f2bf(hi) << 16);
}

__device__ __forceinline__ v8f wmma_bf16(const FragBF& a, const FragBF& b, v8f c) {
    return __builtin_amdgcn_wmma_f32_16x16x32_bf16(false, a.bf, false, b.bf,
                                                   (short)0, c, false, false);
}

// CDNA5 async copy: global -> LDS, 16 bytes per lane, tracked by ASYNCcnt.
__device__ __forceinline__ void async_b128(unsigned lds_byte_addr, unsigned voff_bytes,
                                           unsigned long long sbase) {
    asm volatile("global_load_async_to_lds_b128 %0, %1, %2"
                 :
                 : "v"(lds_byte_addr), "v"(voff_bytes), "s"(sbase)
                 : "memory");
}
__device__ __forceinline__ void wait_async0() {
    asm volatile("s_wait_asynccnt 0x0" ::: "memory");
}
__device__ __forceinline__ unsigned lds_addr_of(const void* p) {
    // LDS aperture: flat addr low 32 bits are the LDS byte offset
    return (unsigned)(unsigned long long)(size_t)p;
}

// ---------------- conversion kernels ----------------
__global__ void k_cvt_x(const float4* __restrict__ x4, uint2* __restrict__ xb2, int n4) {
    int i = blockIdx.x * blockDim.x + threadIdx.x;
    if (i < n4) {
        float4 f = x4[i];
        uint2 o;
        o.x = pack2bf(f.x, f.y);
        o.y = pack2bf(f.z, f.w);
        xb2[i] = o;
    }
}

// LDS-tiled transpose + convert: Wt[n][k] = bf16(W[k][n]); both global sides coalesced.
__global__ __launch_bounds__(256) void k_cvt_wt(const float* __restrict__ w0,
                                                const float* __restrict__ w1,
                                                const float* __restrict__ w2,
                                                const float* __restrict__ w3,
                                                unsigned short* __restrict__ t0,
                                                unsigned short* __restrict__ t1,
                                                unsigned short* __restrict__ t2,
                                                unsigned short* __restrict__ t3) {
    __shared__ unsigned short tile[32][34];
    const float* w;
    unsigned short* t;
    switch (blockIdx.z) {
        case 0: w = w0; t = t0; break;
        case 1: w = w1; t = t1; break;
        case 2: w = w2; t = t2; break;
        default: w = w3; t = t3; break;
    }
    const int tid = threadIdx.x;
    const int k0 = blockIdx.y * 32, n0 = blockIdx.x * 32;
#pragma unroll
    for (int s = 0; s < 4; ++s) {
        int idx = tid + s * 256;
        int i = idx >> 5, j = idx & 31;
        tile[i][j] = f2bf(w[(k0 + i) * 1024 + (n0 + j)]);
    }
    __syncthreads();
    unsigned int* tu = (unsigned int*)t;
#pragma unroll
    for (int s = 0; s < 2; ++s) {
        int idx = tid + s * 256;
        int r = idx >> 4, c = idx & 15;
        unsigned lo = tile[2 * c][r], hv = tile[2 * c + 1][r];
        tu[(n0 + r) * 512 + (k0 >> 1) + c] = lo | (hv << 16);
    }
}

// ---------------- bf16 WMMA GEMM: out[M,N] = A[M,K] * Bt[N,K]^T + bias ----------------
// Double-buffered LDS, async b128 global->LDS staging.
// mode 0: bf16 row-major | mode 1: bf16 per-head-transposed V (Vt[b,h,d,s]) | mode 2: fp32
__global__ __launch_bounds__(256) void k_gemm(const unsigned short* __restrict__ A,
                                              const unsigned short* __restrict__ Bt,
                                              const float* __restrict__ bias,
                                              void* __restrict__ outp, int mode) {
    __shared__ unsigned int As[2][2048]; // 128 rows x 32 bf16 (16 dwords)
    __shared__ unsigned int Bs[2][1024]; // 64 rows  x 32 bf16

    const int tid = threadIdx.x;
    const int lane = tid & 31, l = lane & 15, hi = lane >> 4;
    const int w = tid >> 5;
    const int wm0 = (w >> 1) * 32, wn0 = (w & 1) * 32;
    const int row0 = blockIdx.y * 128;
    const int col0 = blockIdx.x * 64;

    const unsigned asB = lds_addr_of(&As[0][0]);
    const unsigned bsB = lds_addr_of(&Bs[0][0]);
    const unsigned long long aBase = (unsigned long long)(size_t)A;
    const unsigned long long bBase = (unsigned long long)(size_t)Bt;

    // Per K-step staging: A tile = 512 uint4 (2/thread), B tile = 256 uint4 (1/thread)
    const int ra0 = (tid) >> 2, ca0 = (tid) & 3;
    const int ra1 = (tid + 256) >> 2, ca1 = (tid + 256) & 3;
    const int rb = tid >> 2, cb = tid & 3;

#define STAGE(bufi, kb)                                                                     \
    do {                                                                                    \
        async_b128(asB + (bufi) * 8192 + tid * 16,                                          \
                   (unsigned)(((row0 + ra0) * 2048) + (kb) * 2 + ca0 * 16), aBase);         \
        async_b128(asB + (bufi) * 8192 + (tid + 256) * 16,                                  \
                   (unsigned)(((row0 + ra1) * 2048) + (kb) * 2 + ca1 * 16), aBase);         \
        async_b128(bsB + (bufi) * 4096 + tid * 16,                                          \
                   (unsigned)(((col0 + rb) * 2048) + (kb) * 2 + cb * 16), bBase);           \
    } while (0)

    const v8f vzero = {0.f, 0.f, 0.f, 0.f, 0.f, 0.f, 0.f, 0.f};
    v8f acc[2][2];
    acc[0][0] = vzero; acc[0][1] = vzero; acc[1][0] = vzero; acc[1][1] = vzero;

    STAGE(0, 0);
    for (int it = 0; it < 32; ++it) {
        wait_async0();
        __syncthreads();
        if (it + 1 < 32) STAGE((it + 1) & 1, (it + 1) * 32);

        const unsigned int* Ac = &As[it & 1][0];
        const unsigned int* Bc = &Bs[it & 1][0];
        FragBF fa[2], fb[2];
#pragma unroll
        for (int mt = 0; mt < 2; ++mt) {
            const int rowA = wm0 + 16 * mt + l;
#pragma unroll
            for (int v = 0; v < 8; ++v)
                fa[mt].w[v] = Ac[rowA * 16 + 8 * (v >> 2) + 4 * hi + (v & 3)];
        }
#pragma unroll
        for (int nt = 0; nt < 2; ++nt) {
            const int rowB = wn0 + 16 * nt + l;
#pragma unroll
            for (int v = 0; v < 8; ++v)
                fb[nt].w[v] = Bc[rowB * 16 + 8 * (v >> 2) + 4 * hi + (v & 3)];
        }
#pragma unroll
        for (int mt = 0; mt < 2; ++mt)
#pragma unroll
            for (int nt = 0; nt < 2; ++nt)
                acc[mt][nt] = wmma_bf16(fa[mt], fb[nt], acc[mt][nt]);
    }
#undef STAGE

    unsigned short* obf = (unsigned short*)outp;
    float* of = (float*)outp;
#pragma unroll
    for (int mt = 0; mt < 2; ++mt)
#pragma unroll
        for (int nt = 0; nt < 2; ++nt) {
            const int nc = col0 + wn0 + 16 * nt + l;
            const float bs = bias[nc];
#pragma unroll
            for (int r = 0; r < 8; ++r) {
                const int rg = row0 + wm0 + 16 * mt + r + 8 * hi;
                const float val = acc[mt][nt][r] + bs;
                if (mode == 0) {
                    obf[rg * H + nc] = f2bf(val);
                } else if (mode == 1) {
                    const int b_ = rg >> 11, s_ = rg & 2047;
                    const int h_ = nc >> 6, d_ = nc & 63;
                    obf[(((b_ * NH + h_) * HD) + d_) * SQ + s_] = f2bf(val);
                } else {
                    of[rg * H + nc] = val;
                }
            }
        }
}

// ---------------- fused flash attention ----------------
// Block = 8 waves sharing one (batch, head); each wave owns 16 query rows.
// K/V chunks staged into double-buffered LDS via async b128 (1 K + 1 V b128 per thread).
__global__ __launch_bounds__(256) void k_attn(const unsigned short* __restrict__ Q,
                                              const unsigned short* __restrict__ K,
                                              const unsigned short* __restrict__ Vt,
                                              unsigned short* __restrict__ O) {
    __shared__ unsigned int Ks[2][1024];     // 32 kv x 64 bf16 (32 dwords/row)
    __shared__ unsigned int Vs[2][1024];     // 64 d  x 32 bf16 (16 dwords/row)
    __shared__ unsigned short plds[8 * 512]; // per-wave 16x32 bf16 P tile

    const int tid = threadIdx.x;
    const int lane = tid & 31, l = lane & 15, hi = lane >> 4;
    const int wslot = tid >> 5;
    const int hh = (blockIdx.x >> 4) & 15; // block-uniform head
    const int bb = blockIdx.x >> 8;        // block-uniform batch
    const int q0 = (((blockIdx.x << 3) + wslot) & 127) << 4;

    unsigned short* myL = plds + wslot * 512;
    const unsigned int* myLU = (const unsigned int*)plds + wslot * 256;

    const unsigned ksB = lds_addr_of(&Ks[0][0]);
    const unsigned vsB = lds_addr_of(&Vs[0][0]);
    const unsigned long long kBase = (unsigned long long)(size_t)K;
    const unsigned long long vBase = (unsigned long long)(size_t)Vt;

    const int kr = tid >> 3, kc4 = tid & 7; // K: row 0..31, uint4 col 0..7
    const int vd = tid >> 2, vc4 = tid & 3; // V: row d 0..63, uint4 col 0..3

#define STAGE(bufi, kv0)                                                                     \
    do {                                                                                     \
        async_b128(ksB + (bufi) * 4096 + tid * 16,                                           \
                   (unsigned)(((bb * SQ + (kv0) + kr) * 512 + hh * 32 + kc4 * 4) * 4),       \
                   kBase);                                                                   \
        async_b128(vsB + (bufi) * 4096 + tid * 16,                                           \
                   (unsigned)(((((bb * NH + hh) * HD) + vd) * 1024 + ((kv0) >> 1) + vc4 * 4) \
                              * 4),                                                          \
                   vBase);                                                                   \
    } while (0)

    // Q A-fragments, resident for the whole loop (two K=32 fragments over head_dim 64)
    FragBF qf[2];
    {
        const uint4* Qu4 = (const uint4*)Q;
        const int rq4 = (bb * SQ + q0 + l) * 128 + hh * 8;
#pragma unroll
        for (int f = 0; f < 2; ++f) {
            *(uint4*)&qf[f].w[0] = Qu4[rq4 + f * 4 + hi];
            *(uint4*)&qf[f].w[4] = Qu4[rq4 + f * 4 + 2 + hi];
        }
    }

    const v8f vzero = {0.f, 0.f, 0.f, 0.f, 0.f, 0.f, 0.f, 0.f};
    v8f oacc[4];
    oacc[0] = vzero; oacc[1] = vzero; oacc[2] = vzero; oacc[3] = vzero;
    float rmax[8], rsum[8];
#pragma unroll
    for (int r = 0; r < 8; ++r) { rmax[r] = -1e30f; rsum[r] = 0.f; }
    const float scale = 0.125f; // 1/sqrt(64)

    STAGE(0, 0);
    for (int c = 0; c < SQ / 32; ++c) {
        wait_async0();
        __syncthreads();
        if (c + 1 < SQ / 32) STAGE((c + 1) & 1, (c + 1) * 32);

        const unsigned int* Ksb = &Ks[c & 1][0];
        const unsigned int* Vsb = &Vs[c & 1][0];

        // scores S[16q x 32kv] = Q(16x64) @ K^T
        FragBF kf[2][2];
#pragma unroll
        for (int nt = 0; nt < 2; ++nt)
#pragma unroll
            for (int f = 0; f < 2; ++f)
#pragma unroll
                for (int v = 0; v < 8; ++v)
                    kf[nt][f].w[v] = Ksb[(16 * nt + l) * 32 + f * 16 + 8 * (v >> 2) + 4 * hi + (v & 3)];
        v8f s0 = vzero, s1 = vzero;
        s0 = wmma_bf16(qf[0], kf[0][0], s0);
        s0 = wmma_bf16(qf[1], kf[0][1], s0);
        s1 = wmma_bf16(qf[0], kf[1][0], s1);
        s1 = wmma_bf16(qf[1], kf[1][1], s1);

        // online softmax; row m = r + 8*hi lives across this lane-half
        float alpha[8];
#pragma unroll
        for (int r = 0; r < 8; ++r) {
            const float a0 = s0[r] * scale, a1 = s1[r] * scale;
            float t = fmaxf(a0, a1);
            t = fmaxf(t, __shfl_xor(t, 1, 32));
            t = fmaxf(t, __shfl_xor(t, 2, 32));
            t = fmaxf(t, __shfl_xor(t, 4, 32));
            t = fmaxf(t, __shfl_xor(t, 8, 32));
            const float nm = fmaxf(rmax[r], t);
            const float al = __expf(rmax[r] - nm);
            rmax[r] = nm;
            const float e0 = __expf(a0 - nm), e1 = __expf(a1 - nm);
            float ps = e0 + e1;
            ps += __shfl_xor(ps, 1, 32);
            ps += __shfl_xor(ps, 2, 32);
            ps += __shfl_xor(ps, 4, 32);
            ps += __shfl_xor(ps, 8, 32);
            rsum[r] = rsum[r] * al + ps;
            alpha[r] = al;
            const int m = r + 8 * hi;
            myL[m * 32 + l] = f2bf(e0);
            myL[m * 32 + 16 + l] = f2bf(e1);
        }
#pragma unroll
        for (int nd = 0; nd < 4; ++nd)
#pragma unroll
            for (int r = 0; r < 8; ++r) oacc[nd][r] *= alpha[r];

        // wave-local LDS bounce: D-layout P -> A-fragment layout
        asm volatile("s_wait_dscnt 0x0" ::: "memory");
        FragBF pf;
#pragma unroll
        for (int v = 0; v < 8; ++v)
            pf.w[v] = myLU[l * 16 + 8 * (v >> 2) + 4 * hi + (v & 3)];

        // O += P(16x32) @ V(32x64) using LDS-staged transposed V
#pragma unroll
        for (int nd = 0; nd < 4; ++nd) {
            FragBF vf;
#pragma unroll
            for (int v = 0; v < 8; ++v)
                vf.w[v] = Vsb[(nd * 16 + l) * 16 + 8 * (v >> 2) + 4 * hi + (v & 3)];
            oacc[nd] = wmma_bf16(pf, vf, oacc[nd]);
        }
    }
#undef STAGE

#pragma unroll
    for (int r = 0; r < 8; ++r) {
        const float inv = 1.0f / rsum[r];
        const int rg = bb * SQ + q0 + r + 8 * hi;
#pragma unroll
        for (int nd = 0; nd < 4; ++nd)
            O[rg * H + hh * HD + nd * 16 + l] = f2bf(oacc[nd][r] * inv);
    }
}

extern "C" void kernel_launch(void* const* d_in, const int* in_sizes, int n_in,
                              void* d_out, int out_size, void* d_ws, size_t ws_size,
                              hipStream_t stream) {
    const float* x = (const float*)d_in[0];
    const float* Wq = (const float*)d_in[1];
    const float* bq = (const float*)d_in[2];
    const float* Wk = (const float*)d_in[3];
    const float* bk = (const float*)d_in[4];
    const float* Wv = (const float*)d_in[5];
    const float* bv = (const float*)d_in[6];
    const float* Wo = (const float*)d_in[7];
    const float* bo = (const float*)d_in[8];

    char* ws = (char*)d_ws;
    unsigned short* xbf = (unsigned short*)(ws);                  // 8 MiB
    unsigned short* wqt = (unsigned short*)(ws + (8ull << 20));   // 2 MiB each
    unsigned short* wkt = (unsigned short*)(ws + (10ull << 20));
    unsigned short* wvt = (unsigned short*)(ws + (12ull << 20));
    unsigned short* wot = (unsigned short*)(ws + (14ull << 20));
    unsigned short* qb = (unsigned short*)(ws + (16ull << 20));   // 8 MiB
    unsigned short* kbuf = (unsigned short*)(ws + (24ull << 20)); // 8 MiB
    unsigned short* vt = (unsigned short*)(ws + (32ull << 20));   // 8 MiB
    unsigned short* ob = (unsigned short*)(ws + (40ull << 20));   // 8 MiB

    k_cvt_x<<<(M_TOT * H / 4) / 256, 256, 0, stream>>>((const float4*)x, (uint2*)xbf,
                                                       M_TOT * H / 4);
    dim3 gcw(32, 32, 4);
    k_cvt_wt<<<gcw, 256, 0, stream>>>(Wq, Wk, Wv, Wo, wqt, wkt, wvt, wot);

    dim3 gg(H / 64, M_TOT / 128); // 16 x 32 blocks
    k_gemm<<<gg, 256, 0, stream>>>(xbf, wqt, bq, (void*)qb, 0);
    k_gemm<<<gg, 256, 0, stream>>>(xbf, wkt, bk, (void*)kbuf, 0);
    k_gemm<<<gg, 256, 0, stream>>>(xbf, wvt, bv, (void*)vt, 1);

    k_attn<<<(BATCH * NH * (SQ / 16)) / 8, 256, 0, stream>>>(qb, kbuf, vt, ob);

    k_gemm<<<gg, 256, 0, stream>>>(ob, wot, bo, d_out, 2);
}
